// FloatQuantDenseModel_7215545057538
// MI455X (gfx1250) — 8-bit matrix operand order repaired and checked by reading, compiled, not run on hardware
//
#include <hip/hip_runtime.h>
#include <hip/hip_bf16.h>


typedef __attribute__((ext_vector_type(16))) int   v16i;
typedef __attribute__((ext_vector_type(8)))  int   v8i;
typedef __attribute__((ext_vector_type(8)))  float v8f;

#if defined(__has_builtin)
#if __has_builtin(__builtin_amdgcn_cvt_pk_fp8_f32)
#define HAVE_CVT_PK_FP8 1
#endif
#endif

__device__ inline unsigned char fp8_encode_sw(float x) {
  unsigned u = __float_as_uint(x);
  unsigned s = (u >> 24) & 0x80u;
  float ax = fminf(fabsf(x), 448.0f);
  unsigned bits;
  if (ax < 0.015625f) {
    bits = (unsigned)rintf(ax * 512.0f);
  } else {
    unsigned au = __float_as_uint(ax);
    au += 0x7FFFFu + ((au >> 20) & 1u);
    int ef = (int)(au >> 23) - 127 + 7;
    unsigned m = (au >> 20) & 7u;
    if (ef > 15) { ef = 15; m = 6; }
    bits = ((unsigned)ef << 3) | m;
  }
  return (unsigned char)(s | bits);
}

template <bool WORD>
__device__ inline int cvt2fp8(float a, float b, int old) {
#ifdef HAVE_CVT_PK_FP8
  return __builtin_amdgcn_cvt_pk_fp8_f32(a, b, old, WORD);
#else
  unsigned p = (unsigned)fp8_encode_sw(a) | ((unsigned)fp8_encode_sw(b) << 8);
  return WORD ? (int)(((unsigned)old & 0x0000FFFFu) | (p << 16))
              : (int)(((unsigned)old & 0xFFFF0000u) | p);
#endif
}

__device__ inline unsigned char fp8_encode1(float a) {
  return (unsigned char)((unsigned)cvt2fp8<false>(a, 0.0f, 0) & 0xFFu);
}

#define IN_K   1024
#define DW     58
#define NC     12
#define NPAD   64
#define BN_EPS 1e-5f

__global__ void prep_kernel(const float* __restrict__ W1,
                            const float* __restrict__ W2,
                            const float* __restrict__ gamma,
                            const float* __restrict__ beta,
                            const float* __restrict__ mean,
                            const float* __restrict__ var,
                            unsigned char* __restrict__ w1q8,
                            float* __restrict__ bnInv,
                            float* __restrict__ bnBias,
                            unsigned char* __restrict__ w2frag) {
  const int tid = threadIdx.x + blockIdx.x * blockDim.x;
  const int nthreads = blockDim.x * gridDim.x;
  for (int i = tid; i < NPAD * IN_K; i += nthreads) {
    const int n = i >> 10, k = i & (IN_K - 1);
    const float v = (n < DW) ? W1[n * IN_K + k] : 0.0f;
    w1q8[i] = fp8_encode1(v);
  }
  if (blockIdx.x == 0) {
    for (int n = threadIdx.x; n < NPAD; n += blockDim.x) {
      if (n < DW) {
        const float inv = gamma[n] * rsqrtf(var[n] + BN_EPS);
        bnInv[n]  = inv;
        bnBias[n] = beta[n] - mean[n] * inv;
      } else {
        bnInv[n] = 0.0f; bnBias[n] = 0.0f;
      }
    }
    for (int idx = threadIdx.x; idx < 32 * 32; idx += blockDim.x) {
      const int lane = idx >> 5, j = idx & 31;
      const int n  = lane & 15;
      const int hi = lane >> 4;
      const int k  = (j & 15) + 16 * hi + 32 * (j >> 4);
      const float v = (n < NC && k < DW) ? W2[n * DW + k] : 0.0f;
      w2frag[idx] = fp8_encode1(v);
    }
  }
}

__global__ __launch_bounds__(256)
void fused_mlp_kernel(const float* __restrict__ x,
                      const unsigned char* __restrict__ w1q8,
                      const float* __restrict__ bnInv,
                      const float* __restrict__ bnBias,
                      const unsigned char* __restrict__ w2frag,
                      const float* __restrict__ act_scale,
                      float* __restrict__ out) {
  __shared__ unsigned char aS[8][16 * 72 + 16];

  const int lane = threadIdx.x & 31;
  const int wave = threadIdx.x >> 5;
  const int lo   = lane & 15;
  const int hi   = lane >> 4;
  const int m    = lo;
  const int rowBase = blockIdx.x * 128 + wave * 16;

  const float s_act = act_scale[0];
  const float inv_s = 1.0f / s_act;

  float binv[4], bb[4];
#pragma unroll
  for (int t = 0; t < 4; ++t) { binv[t] = bnInv[16 * t + lo]; bb[t] = bnBias[16 * t + lo]; }

  const v8i b2 = *(const v8i*)(w2frag + lane * 32);

  v8f acc[4] = {v8f{}, v8f{}, v8f{}, v8f{}};
  const float* xrow = x + (size_t)(rowBase + m) * IN_K;

  for (int c = 0; c < 8; ++c) {
    const int k0 = c * 128;
    if (c < 7) __builtin_prefetch(xrow + k0 + 128 + 8 * hi, 0, 3);

    v16i afrag;
#pragma unroll
    for (int g = 0; g < 8; ++g) {
      const float4 f0 = *(const float4*)(xrow + k0 + 32 * (g >> 1) + 16 * hi + 8 * (g & 1));
      const float4 f1 = *(const float4*)(xrow + k0 + 32 * (g >> 1) + 16 * hi + 8 * (g & 1) + 4);
      int p0 = 0, p1 = 0;
      p0 = cvt2fp8<false>(f0.x, f0.y, p0);
      p0 = cvt2fp8<true >(f0.z, f0.w, p0);
      p1 = cvt2fp8<false>(f1.x, f1.y, p1);
      p1 = cvt2fp8<true >(f1.z, f1.w, p1);
      afrag[2 * g]     = p0;
      afrag[2 * g + 1] = p1;
    }

#pragma unroll
    for (int t = 0; t < 4; ++t) {
      const unsigned char* bp = w1q8 + (size_t)(16 * t + lo) * IN_K + k0 + 16 * hi;
      v16i bfrag;
#pragma unroll
      for (int q = 0; q < 4; ++q) {
        const int4 d = *(const int4*)(bp + 32 * q);
        bfrag[4 * q + 0] = d.x; bfrag[4 * q + 1] = d.y;
        bfrag[4 * q + 2] = d.z; bfrag[4 * q + 3] = d.w;
      }
      acc[t] = __builtin_amdgcn_wmma_f32_16x16x128_fp8_fp8(
          afrag, bfrag, (short)0, acc[t], false, false);
    }
  }

  unsigned char* aw = &aS[wave][0];
#pragma unroll
  for (int t = 0; t < 4; ++t) {
#pragma unroll
    for (int r = 0; r < 8; ++r) {
      const float h = acc[t][r];
      const float a = fmaxf(h * binv[t] + bb[t], 0.0f);
      const float q = fminf(rintf(a * inv_s), 255.0f) * s_act;
      aw[(r + 8 * hi) * 72 + 16 * t + lo] = fp8_encode1(q);
    }
  }
  asm volatile("s_wait_dscnt 0x0" ::: "memory");

  v8i a2;
  const unsigned char* ap = aw + m * 72 + 16 * hi;
#pragma unroll
  for (int g = 0; g < 4; ++g) {
    const int2 d = *(const int2*)(ap + 32 * (g >> 1) + 8 * (g & 1));
    a2[2 * g] = d.x; a2[2 * g + 1] = d.y;
  }
  v8f o = {};
  o = __builtin_amdgcn_wmma_f32_16x16x64_fp8_fp8(a2, b2, (short)0, o, false, false);

  if (lo < NC) {
#pragma unroll
    for (int r = 0; r < 8; ++r) {
      out[(size_t)(rowBase + r + 8 * hi) * NC + lo] = o[r];
    }
  }
}

extern "C" void kernel_launch(void* const* d_in, const int* in_sizes, int n_in,
                              void* d_out, int out_size, void* d_ws, size_t ws_size,
                              hipStream_t stream) {
  const float* x         = (const float*)d_in[0];
  const float* W1        = (const float*)d_in[1];
  const float* W2        = (const float*)d_in[2];
  const float* bn_gamma  = (const float*)d_in[3];
  const float* bn_beta   = (const float*)d_in[4];
  const float* bn_mean   = (const float*)d_in[5];
  const float* bn_var    = (const float*)d_in[6];
  const float* act_scale = (const float*)d_in[7];
  float* out = (float*)d_out;

  unsigned char* ws     = (unsigned char*)d_ws;
  unsigned char* w1q8   = ws;
  float*         bnInv  = (float*)(ws + 65536);
  float*         bnBias = (float*)(ws + 65536 + 256);
  unsigned char* w2frag = ws + 65536 + 512;

  hipLaunchKernelGGL(prep_kernel, dim3(64), dim3(256), 0, stream,
                     W1, W2, bn_gamma, bn_beta, bn_mean, bn_var,
                     w1q8, bnInv, bnBias, w2frag);

  const int rows   = in_sizes[0] / IN_K;
  const int blocks = rows / 128;
  hipLaunchKernelGGL(fused_mlp_kernel, dim3(blocks), dim3(256), 0, stream,
                     x, w1q8, bnInv, bnBias, w2frag, act_scale, out);

  (void)n_in; (void)out_size; (void)ws_size;
}
